// DenseSAKELayer_76081050681987
// MI455X (gfx1250) — matrix operand order repaired and checked by reading, not compiled, not run on hardware
//
#include <hip/hip_runtime.h>
#include <hip/hip_bf16.h>

#define B_ 4
#define N_ 512
#define F_ 64
#define H_ 64
#define C_ 32
#define NWAVE 8
#define TS 88   // f16 tile row stride: 176B rows -> 16B aligned b128 reads, no half-wave bank conflict

typedef _Float16 v16h __attribute__((ext_vector_type(16)));
typedef float    v8f  __attribute__((ext_vector_type(8)));
typedef unsigned uu4  __attribute__((ext_vector_type(4)));

// Fast silu: v_exp_f32 + v_rcp_f32 (activations are truncated to f16 right after,
// so the ~1ulp hardware rcp is plenty; avoids IEEE div_scale/div_fmas microcode).
__device__ __forceinline__ float silu_f(float v) {
    return v * __builtin_amdgcn_rcpf(1.0f + __expf(-v));
}

union FragU { v16h v; uu4 q[2]; };

// Weight B-fragment (32x16 f16), pre-packed in LDS as [frag][lane][vgpr] u32.
__device__ __forceinline__ v16h load_bfrag(const unsigned* sFrag, int f, int lane) {
    FragU r;
    const uu4* p = (const uu4*)(sFrag + f * 256 + lane * 8);
    r.q[0] = p[0];
    r.q[1] = p[1];
    return r.v;
}

// A-fragment (16x32 f16) from row-major f16 tile [16][TS].
// lane: M = lane&15; lanes<16 hold K base 0, lanes>=16 base 8; v0..3 = K base..base+7, v4..7 = +16.
__device__ __forceinline__ v16h load_afrag(const _Float16* tile, int lane, int kt) {
    const int M = lane & 15;
    const int base = (lane < 16) ? 0 : 8;
    const _Float16* p = tile + M * TS + kt * 32 + base;
    FragU r;
    r.q[0] = *(const uu4*)(p);
    r.q[1] = *(const uu4*)(p + 16);
    return r.v;
}

// ---------- prep: P = h @ We1[0:64]  (j-side),  Q = h @ We1[64:128] + be1  (i-side) ----------
__global__ __launch_bounds__(256) void sake_prep(
    const float* __restrict__ h, const float* __restrict__ We1, const float* __restrict__ be1,
    float* __restrict__ P, float* __restrict__ Q) {
    int idx = blockIdx.x * 256 + threadIdx.x;       // [0, B*N*F)
    int f = idx & 63;
    const float* hrow = h + (size_t)(idx >> 6) * 64;
    float p = 0.0f, q = be1[f];
#pragma unroll 8
    for (int k = 0; k < 64; ++k) {
        float hv = hrow[k];
        p += hv * We1[k * 64 + f];
        q += hv * We1[(64 + k) * 64 + f];
    }
    P[idx] = p;
    Q[idx] = q;
}

// ---------- main: one wave per (b, i); j-chunks of 16; WMMA pipeline ----------
__global__ __launch_bounds__(256) void sake_main(
    const float* __restrict__ h, const float* __restrict__ x, const float* __restrict__ mask,
    const float* __restrict__ We1,
    const float* __restrict__ We2, const float* __restrict__ be2,
    const float* __restrict__ Wc1, const float* __restrict__ bc1,
    const float* __restrict__ Wc2, const float* __restrict__ bc2,
    const float* __restrict__ Wp1, const float* __restrict__ bp1,
    const float* __restrict__ Wp2, const float* __restrict__ bp2,
    const float* __restrict__ Wn1, const float* __restrict__ bn1,
    const float* __restrict__ Wn2, const float* __restrict__ bn2,
    const float* __restrict__ P, const float* __restrict__ Q,
    float* __restrict__ out) {

    __shared__ unsigned sFrag[20 * 256];                      // 20 B-fragments, packed
    __shared__ float4   sVec[NWAVE][16];                      // per chunk: (vec.xyz, mask)
    __shared__ alignas(16) _Float16 sTile[NWAVE][16 * TS];    // layout-conversion tile
    __shared__ float    sCN[NWAVE][32];
    __shared__ float    sT1[NWAVE][64];
    __shared__ float    sNI[NWAVE][192];
    __shared__ float    sT2[NWAVE][64];

    // ---- pack weight fragments into LDS (f16), fragment-native order ----
    for (int t = threadIdx.x; t < 20 * 256; t += 256) {
        int f = t >> 8, rem = t & 255, l = rem >> 3, v = rem & 7;
        const float* W; int ldn, kt, nt;
        if (f < 8)       { W = We2; ldn = 64; kt = f >> 2;        nt = f & 3; }
        else if (f < 16) { W = Wc1; ldn = 64; kt = (f - 8) >> 2;  nt = (f - 8) & 3; }
        else             { W = Wc2; ldn = 32; kt = (f - 16) >> 1; nt = (f - 16) & 1; }
        int K = kt * 32 + ((l < 16) ? 0 : 8) + 2 * v + ((v < 4) ? 0 : 8);
        int Nc = nt * 16 + (l & 15);
        union { _Float16 hh[2]; unsigned u; } pk;
        pk.hh[0] = (_Float16)W[K * ldn + Nc];
        pk.hh[1] = (_Float16)W[(K + 1) * ldn + Nc];
        sFrag[t] = pk.u;
    }
    __syncthreads();

    const int wv = threadIdx.x >> 5;
    const int lane = threadIdx.x & 31;
    const int g = blockIdx.x * NWAVE + wv;
    const int b = g >> 9;
    const int i = g & 511;
    const size_t rowbi = (size_t)(b * N_ + i);
    const int cl = lane & 15;
    const int mh = (lane >> 4) << 3;      // 0 or 8: this half-wave's M offset in C layout

    const float qi0 = Q[rowbi * 64 + lane];
    const float qi1 = Q[rowbi * 64 + lane + 32];
    const float w0 = We1[128 * 64 + lane];
    const float w1 = We1[128 * 64 + lane + 32];
    const float xi0 = x[rowbi * 3 + 0];
    const float xi1 = x[rowbi * 3 + 1];
    const float xi2 = x[rowbi * 3 + 2];

    float be2r[4], bc1r[4];
#pragma unroll
    for (int nt = 0; nt < 4; ++nt) { be2r[nt] = be2[nt * 16 + cl]; bc1r[nt] = bc1[nt * 16 + cl]; }
    float bc2r[2] = { bc2[cl], bc2[16 + cl] };

    float he[4] = {0.f, 0.f, 0.f, 0.f};
    float cb[2][3] = {{0.f,0.f,0.f},{0.f,0.f,0.f}};
    _Float16* tile = &sTile[wv][0];
    const float* Prow = P + (size_t)b * N_ * 64;
    const float* mrow = mask + rowbi * N_;

    for (int jc = 0; jc < 32; ++jc) {
        __syncthreads();   // block-uniform; also fences LDS so frags reload (keeps VGPRs < 256)
        const int j0 = jc * 16;

        // all lanes compute n2 for j = j0 + (lane&15); lanes>=16 duplicate lanes<16
        const int j = j0 + cl;
        float dx = x[(size_t)(b * N_ + j) * 3 + 0] - xi0;
        float dy = x[(size_t)(b * N_ + j) * 3 + 1] - xi1;
        float dz = x[(size_t)(b * N_ + j) * 3 + 2] - xi2;
        float n2 = dx * dx + dy * dy + dz * dz;
        if (lane < 16) {
            float inv = __builtin_amdgcn_rcpf(n2 * n2 + 1e-10f);
            sVec[wv][lane] = make_float4(dx * inv, dy * inv, dz * inv, mrow[j]);
        }
        __builtin_prefetch(Prow + (size_t)(((jc + 1) & 31) * 16) * 64, 0, 1);

        // pre-activation edge layer 1: P[j] + Q[i] + n2 * We1[128]; silu -> f16 tile
#pragma unroll
        for (int t = 0; t < 32; ++t) {
            int M = t >> 1;
            int fo = (t & 1) << 5;
            float n2b = __shfl(n2, M, 32);          // uniform src lane -> v_readlane broadcast
            float pj = Prow[(size_t)(j0 + M) * 64 + fo + lane];
            float val = pj + ((t & 1) ? qi1 : qi0) + n2b * ((t & 1) ? w1 : w0);
            tile[M * TS + fo + lane] = (_Float16)silu_f(val);
        }

        float mr[8];
#pragma unroll
        for (int r = 0; r < 8; ++r) mr[r] = sVec[wv][r + mh].w;

        // ---- stage 1: @ We2 (K=64) ----
        v16h a0 = load_afrag(tile, lane, 0);
        v16h a1 = load_afrag(tile, lane, 1);
        v8f acc[4];
#pragma unroll
        for (int nt = 0; nt < 4; ++nt) {
            v8f c;
#pragma unroll
            for (int r = 0; r < 8; ++r) c[r] = be2r[nt];
            c = __builtin_amdgcn_wmma_f32_16x16x32_f16(false, a0, false, load_bfrag(sFrag, nt, lane),     (short)0, c, false, false);
            c = __builtin_amdgcn_wmma_f32_16x16x32_f16(false, a1, false, load_bfrag(sFrag, 4 + nt, lane), (short)0, c, false, false);
            acc[nt] = c;
        }
        // silu -> h_e_mtx; masked h_e accumulate; write back for stage-2 A
#pragma unroll
        for (int nt = 0; nt < 4; ++nt) {
#pragma unroll
            for (int r = 0; r < 8; ++r) {
                float s = silu_f(acc[nt][r]);
                he[nt] += s * mr[r];
                tile[(r + mh) * TS + nt * 16 + cl] = (_Float16)s;
            }
        }

        // ---- stage 2: @ Wc1 (K=64) ----
        a0 = load_afrag(tile, lane, 0);
        a1 = load_afrag(tile, lane, 1);
#pragma unroll
        for (int nt = 0; nt < 4; ++nt) {
            v8f c;
#pragma unroll
            for (int r = 0; r < 8; ++r) c[r] = bc1r[nt];
            c = __builtin_amdgcn_wmma_f32_16x16x32_f16(false, a0, false, load_bfrag(sFrag, 8 + nt, lane),  (short)0, c, false, false);
            c = __builtin_amdgcn_wmma_f32_16x16x32_f16(false, a1, false, load_bfrag(sFrag, 12 + nt, lane), (short)0, c, false, false);
#pragma unroll
            for (int r = 0; r < 8; ++r)
                tile[(r + mh) * TS + nt * 16 + cl] = (_Float16)silu_f(c[r]);
        }

        // ---- stage 3: coeff = @ Wc2 (K=64, N=32) ----
        a0 = load_afrag(tile, lane, 0);
        a1 = load_afrag(tile, lane, 1);
        v8f cf[2];
#pragma unroll
        for (int nt = 0; nt < 2; ++nt) {
            v8f c;
#pragma unroll
            for (int r = 0; r < 8; ++r) c[r] = bc2r[nt];
            c = __builtin_amdgcn_wmma_f32_16x16x32_f16(false, a0, false, load_bfrag(sFrag, 16 + nt, lane), (short)0, c, false, false);
            c = __builtin_amdgcn_wmma_f32_16x16x32_f16(false, a1, false, load_bfrag(sFrag, 18 + nt, lane), (short)0, c, false, false);
            cf[nt] = c;
        }
        // comb_sum[c][d] += coeff * vec * mask
#pragma unroll
        for (int r = 0; r < 8; ++r) {
            float4 vm = sVec[wv][r + mh];
#pragma unroll
            for (int nt = 0; nt < 2; ++nt) {
                float cm = cf[nt][r] * vm.w;
                cb[nt][0] += cm * vm.x;
                cb[nt][1] += cm * vm.y;
                cb[nt][2] += cm * vm.z;
            }
        }
    }

    // ---- cross-half reduction (wave32: xor 16 swaps C-layout halves) ----
#pragma unroll
    for (int nt = 0; nt < 4; ++nt) he[nt] += __shfl_xor(he[nt], 16, 32);
#pragma unroll
    for (int nt = 0; nt < 2; ++nt)
#pragma unroll
        for (int d = 0; d < 3; ++d) cb[nt][d] += __shfl_xor(cb[nt][d], 16, 32);

    float cn0 = cb[0][0]*cb[0][0] + cb[0][1]*cb[0][1] + cb[0][2]*cb[0][2];
    float cn1 = cb[1][0]*cb[1][0] + cb[1][1]*cb[1][1] + cb[1][2]*cb[1][2];
    if (lane < 16) { sCN[wv][lane] = cn0; sCN[wv][16 + lane] = cn1; }

    // ---- post_norm_mlp: silu(cn @ Wp1 + bp1) @ Wp2 + bp2 ----
    float t1a = bp1[lane], t1b = bp1[lane + 32];
    for (int c = 0; c < 32; ++c) {
        float cv = sCN[wv][c];
        t1a += cv * Wp1[c * 64 + lane];
        t1b += cv * Wp1[c * 64 + lane + 32];
    }
    sT1[wv][lane] = silu_f(t1a);
    sT1[wv][lane + 32] = silu_f(t1b);
    float hca = bp2[lane], hcb = bp2[lane + 32];
    for (int k = 0; k < 64; ++k) {
        float tv = sT1[wv][k];
        hca += tv * Wp2[k * 64 + lane];
        hcb += tv * Wp2[k * 64 + lane + 32];
    }

    // ---- node_mlp on [h, h_e, h_comb]; residual add ----
    float ha = h[rowbi * 64 + lane], hb = h[rowbi * 64 + lane + 32];
    sNI[wv][lane] = ha;
    sNI[wv][lane + 32] = hb;
    if (lane < 16) {
#pragma unroll
        for (int nt = 0; nt < 4; ++nt) sNI[wv][64 + nt * 16 + lane] = he[nt];
    }
    sNI[wv][128 + lane] = hca;
    sNI[wv][160 + lane] = hcb;

    float n1a = bn1[lane], n1b = bn1[lane + 32];
    for (int k = 0; k < 192; ++k) {
        float v = sNI[wv][k];
        n1a += v * Wn1[k * 64 + lane];
        n1b += v * Wn1[k * 64 + lane + 32];
    }
    sT2[wv][lane] = silu_f(n1a);
    sT2[wv][lane + 32] = silu_f(n1b);
    float oa = bn2[lane] + ha, ob = bn2[lane + 32] + hb;
    for (int k = 0; k < 64; ++k) {
        float tv = sT2[wv][k];
        oa += tv * Wn2[k * 64 + lane];
        ob += tv * Wn2[k * 64 + lane + 32];
    }
    out[rowbi * 64 + lane] = oa;
    out[rowbi * 64 + lane + 32] = ob;
}

extern "C" void kernel_launch(void* const* d_in, const int* in_sizes, int n_in,
                              void* d_out, int out_size, void* d_ws, size_t ws_size,
                              hipStream_t stream) {
    const float* h    = (const float*)d_in[0];
    const float* x    = (const float*)d_in[1];
    const float* mask = (const float*)d_in[2];
    const float* We1  = (const float*)d_in[3];
    const float* be1  = (const float*)d_in[4];
    const float* We2  = (const float*)d_in[5];
    const float* be2  = (const float*)d_in[6];
    const float* Wc1  = (const float*)d_in[7];
    const float* bc1  = (const float*)d_in[8];
    const float* Wc2  = (const float*)d_in[9];
    const float* bc2  = (const float*)d_in[10];
    const float* Wp1  = (const float*)d_in[11];
    const float* bp1  = (const float*)d_in[12];
    const float* Wp2  = (const float*)d_in[13];
    const float* bp2  = (const float*)d_in[14];
    const float* Wn1  = (const float*)d_in[15];
    const float* bn1  = (const float*)d_in[16];
    const float* Wn2  = (const float*)d_in[17];
    const float* bn2  = (const float*)d_in[18];
    float* out = (float*)d_out;

    float* P = (float*)d_ws;
    float* Q = P + (size_t)B_ * N_ * F_;

    sake_prep<<<(B_ * N_ * F_) / 256, 256, 0, stream>>>(h, We1, be1, P, Q);
    sake_main<<<(B_ * N_) / NWAVE, 256, 0, stream>>>(
        h, x, mask, We1, We2, be2, Wc1, bc1, Wc2, bc2,
        Wp1, bp1, Wp2, bp2, Wn1, bn1, Wn2, bn2, P, Q, out);

    // x passes through unchanged -> second tuple output
    hipMemcpyAsync(out + (size_t)B_ * N_ * F_, x, (size_t)B_ * N_ * 3 * sizeof(float),
                   hipMemcpyDeviceToDevice, stream);
}
